// XposMultiheadAttention2D_68032281969113
// MI455X (gfx1250) — matrix operand order repaired and checked by reading, not compiled, not run on hardware
//
#include <hip/hip_runtime.h>
#include <hip/hip_bf16.h>

// ---------------------------------------------------------------------------
// Types for CDNA5 WMMA f16 16x16x32 (wave32)
// ---------------------------------------------------------------------------
typedef _Float16 v16h __attribute__((ext_vector_type(16)));
typedef _Float16 v8h  __attribute__((ext_vector_type(8)));
typedef float    v8f  __attribute__((ext_vector_type(8)));

union HV16 { v16h v; v8h h[2]; };

#define WMMA_F16(a, b, c) \
  __builtin_amdgcn_wmma_f32_16x16x32_f16(false, (a), false, (b), (short)0, (c), false, false)

// DPP cross-lane exchange within 16-lane rows (VALU-only, no ds_bpermute).
#define DPP_XOR(x, ctrl) \
  __int_as_float(__builtin_amdgcn_update_dpp(0, __float_as_int(x), (ctrl), 0xf, 0xf, true))

__device__ __forceinline__ float row16_max(float x) {
  x = fmaxf(x, DPP_XOR(x, 0xB1));   // quad_perm [1,0,3,2]  : lane ^ 1
  x = fmaxf(x, DPP_XOR(x, 0x4E));   // quad_perm [2,3,0,1]  : lane ^ 2
  x = fmaxf(x, DPP_XOR(x, 0x141));  // row_half_mirror      : lane ^ 7 (quad-reduced -> xor4)
  x = fmaxf(x, DPP_XOR(x, 0x140));  // row_mirror           : lane ^ 15 (-> xor8)
  return x;
}
__device__ __forceinline__ float row16_sum(float x) {
  x += DPP_XOR(x, 0xB1);
  x += DPP_XOR(x, 0x4E);
  x += DPP_XOR(x, 0x141);
  x += DPP_XOR(x, 0x140);
  return x;
}

// ---------------------------------------------------------------------------
// Async global -> LDS copy (CDNA5 ASYNCcnt path), with portable fallback.
// Builtin signature (from hipcc diagnostic): param0 = v4i32 AS1*, param1 = AS3*.
// ---------------------------------------------------------------------------
#if __has_builtin(__builtin_amdgcn_global_load_async_to_lds_b128)
typedef int async_v4i __attribute__((vector_size(16)));
typedef __attribute__((address_space(1))) async_v4i* as1_v4ip;
typedef __attribute__((address_space(3))) async_v4i* as3_v4ip;
#define TILE_COPY16(dst, src) \
  __builtin_amdgcn_global_load_async_to_lds_b128((as1_v4ip)(src), (as3_v4ip)(dst), 0, 0)
#if __has_builtin(__builtin_amdgcn_s_wait_asynccnt)
#define WAIT_ASYNC(n) __builtin_amdgcn_s_wait_asynccnt(n)
#else
#define WAIT_ASYNC(n) asm volatile("s_wait_asynccnt %0" ::"n"(n) : "memory")
#endif
#else
// Fallback: regular load + ds_store; __syncthreads' implicit waitcnts order it.
#define TILE_COPY16(dst, src) (*(v8h*)(dst) = *(const v8h*)(src))
#define WAIT_ASYNC(n) ((void)0)
#endif

static constexpr int T_TOK   = 4608;   // 2 * 48 * 48 tokens
static constexpr int EMB     = 1024;
static constexpr int NHEAD   = 16;
static constexpr int HDIM    = 64;
static constexpr int SEQ     = 2304;   // 48*48 per image
static constexpr int BHEADS  = 32;     // 2 * 16
static constexpr int NWIN    = SEQ / 32;  // 72 key windows

// ---------------------------------------------------------------------------
// f32 -> f16 convert
// ---------------------------------------------------------------------------
__global__ void f32_to_f16_kernel(const float* __restrict__ in,
                                  _Float16* __restrict__ out, int n) {
  int i = blockIdx.x * blockDim.x + threadIdx.x;
  if (i < n) out[i] = (_Float16)in[i];
}

// ---------------------------------------------------------------------------
// xPos tables: [48 positions][16 pairs], q (scale) and k (1/scale) variants
// ---------------------------------------------------------------------------
__global__ void xpos_tables_kernel(float* __restrict__ sq, float* __restrict__ cq,
                                   float* __restrict__ sk, float* __restrict__ ck) {
  int id = blockIdx.x * blockDim.x + threadIdx.x;
  if (id >= 48 * 16) return;
  int p = id >> 4, i = id & 15;
  float sb    = (2.0f * (float)i + 0.4f * 32.0f) / (1.4f * 32.0f);
  float power = ((float)p - 24.0f) / 512.0f;          // min_pos = -24
  float scale = __powf(sb, power);
  float invf  = __powf(10000.0f, -(float)i / 16.0f);
  float ang   = (float)p * invf;
  float s = __sinf(ang), c = __cosf(ang);
  sq[id] = s * scale;  cq[id] = c * scale;            // downscale=False (q)
  sk[id] = s / scale;  ck[id] = c / scale;            // downscale=True  (k)
}

// ---------------------------------------------------------------------------
// GEMM: out[t,o] = sum_e X[t,e]*W[o,e] + bias[o]  (X:[4608,1024] f16, W:[1024,1024] f16)
// Per wave: 16x64 output tile (M tile x one full head of N).
// MODE 0: f32 out (+bias) -> outF                        (output projection)
// MODE 1: Q: *0.125, xPos rotate -> outQK [32][2304][64] f16
// MODE 2: K: xPos rotate (downscaled tables) -> outQK
// MODE 3: V: -> outVt [32][64][2304] f16 (transposed for PV B-matrix reads)
// ---------------------------------------------------------------------------
template <int MODE>
__global__ __launch_bounds__(256)
void proj_gemm_kernel(const _Float16* __restrict__ X, const _Float16* __restrict__ W,
                      const float* __restrict__ bias,
                      float* __restrict__ outF,
                      _Float16* __restrict__ outQK,
                      _Float16* __restrict__ outVt,
                      const float* __restrict__ sinT, const float* __restrict__ cosT) {
  const int wave = threadIdx.x >> 5;
  const int lane = threadIdx.x & 31;
  const int lm   = lane & 15;
  const int hiK  = lane >> 4;                 // 0 or 1
  const int task  = blockIdx.x * 8 + wave;    // 288 * 16 = 4608 tasks
  const int mtile = task >> 4;
  const int strip = task & 15;                // == head for this N strip
  const int m0 = mtile * 16, n0 = strip * 64;

  // A layout: lane -> row m0+lm; contiguous 16 K at hiK*16
  const _Float16* arow = X + (size_t)(m0 + lm) * EMB + hiK * 16;

  v8f acc[4] = {};
  for (int kc = 0; kc < EMB; kc += 32) {
    HV16 a;
    a.h[0] = *(const v8h*)(arow + kc);
    a.h[1] = *(const v8h*)(arow + kc + 8);
#pragma unroll
    for (int nn = 0; nn < 4; ++nn) {
      // B layout: lane -> N = lm, contiguous 16 K at kc + hiK*16; B[k,n] = W[n,k]
      HV16 b;
      b.v = *(const v16h*)(W + (size_t)(n0 + nn * 16 + lm) * EMB + kc + hiK * 16);
      acc[nn] = WMMA_F16(a.v, b.v, acc[nn]);
    }
  }

#pragma unroll
  for (int nn = 0; nn < 4; ++nn) {
#pragma unroll
    for (int r = 0; r < 8; ++r) {
      const int e   = n0 + nn * 16 + lm;
      const int row = r + hiK * 8;
      const int t   = m0 + row;
      float x = acc[nn][r] + bias[e];
      if (MODE == 0) {
        outF[(size_t)t * EMB + e] = x;
      } else if (MODE == 3) {
        const int b = t / SEQ, s = t - b * SEQ;
        const int d = nn * 16 + lm;
        const int bh = b * NHEAD + strip;
        outVt[((size_t)bh * HDIM + d) * SEQ + s] = (_Float16)x;
      } else {
        if (MODE == 1) x *= 0.125f;            // HEAD_DIM^-0.5 (commutes with rotation)
        const float xo = DPP_XOR(x, 0xB1);     // rotation partner = adjacent N column
        const int b = t / SEQ, s = t - b * SEQ;
        const int hh = s / 48, ww = s - hh * 48;
        const int d = nn * 16 + lm;
        const int pos = (d < 32) ? ww : hh;    // first half rotates along W, second along H
        const int pi  = (d & 31) >> 1;
        const float sv = sinT[pos * 16 + pi];
        const float cv = cosT[pos * 16 + pi];
        const float res = x * cv + ((lane & 1) ? xo * sv : -xo * sv);
        const int bh = b * NHEAD + strip;
        outQK[((size_t)bh * SEQ + s) * HDIM + d] = (_Float16)res;
      }
    }
  }
}

// ---------------------------------------------------------------------------
// Flash attention: one block (8 waves) per (batch-head, 128-query slab).
//  - 32-key K/V tiles staged ONCE per block in LDS (8x less L2 traffic),
//    double-buffered with async global->LDS copies (ASYNCcnt) overlapping
//    the next window's fill with the current window's math
//  - row max reduced with DPP16 (VALU only)
//  - row sum computed by the matrix unit via a register-built "ones column"
//    B-tile accumulated into O[4]; the alpha rescale maintains l for free
//  - P staged via wave-private LDS (stride 40 f16 rows) for C->A layout
// ---------------------------------------------------------------------------
__global__ __launch_bounds__(256)
void attention_kernel(const _Float16* __restrict__ Qh, const _Float16* __restrict__ Kh,
                      const _Float16* __restrict__ Vt, _Float16* __restrict__ Aout) {
  __shared__ __attribute__((aligned(16))) _Float16 kbuf[2][32][72]; // 144B rows
  __shared__ __attribute__((aligned(16))) _Float16 vbuf[2][64][40]; // 80B rows
  __shared__ __attribute__((aligned(16))) _Float16 plds[8][16][40];

  const int tid  = threadIdx.x;
  const int wave = tid >> 5;
  const int lane = tid & 31;
  const int lm   = lane & 15;
  const int hiK  = lane >> 4;
  const int bh   = blockIdx.x / 18;           // 32 heads x 18 query slabs
  const int qt   = (blockIdx.x % 18) * 8 + wave;

  // Cooperative tile fill: 256 threads x 16B per tile for K and for V.
  const _Float16* kgb = Kh + ((size_t)bh * SEQ + (tid >> 3)) * HDIM + (tid & 7) * 8;
  const _Float16* vgb = Vt + ((size_t)bh * HDIM + (tid >> 2)) * SEQ + (tid & 3) * 8;
  _Float16* klp = &kbuf[0][tid >> 3][(tid & 7) * 8];
  _Float16* vlp = &vbuf[0][tid >> 2][(tid & 3) * 8];

#define FILL_TILES(buf, key0)                                          \
  do {                                                                 \
    TILE_COPY16(klp + (buf) * (32 * 72), kgb + (size_t)(key0) * HDIM); \
    TILE_COPY16(vlp + (buf) * (64 * 40), vgb + (key0));                \
  } while (0)

  // Q tile 16x64 -> two A matrices (K chunks 0-31, 32-63)
  HV16 qa[2];
  const _Float16* qrow = Qh + ((size_t)bh * SEQ + qt * 16 + lm) * HDIM + hiK * 16;
#pragma unroll
  for (int kk = 0; kk < 2; ++kk) {
    qa[kk].h[0] = *(const v8h*)(qrow + 32 * kk);
    qa[kk].h[1] = *(const v8h*)(qrow + 32 * kk + 8);
  }

  // Register-built "ones column" B tile: B[k,0]=1, B[k,n!=0]=0 (no memory reads)
  HV16 bones;
  {
    const _Float16 one = (lm == 0) ? (_Float16)1.0f : (_Float16)0.0f;
#pragma unroll
    for (int i = 0; i < 16; ++i) bones.v[i] = one;
  }

  float m_i[8];
  v8f O[5] = {};                               // O[0..3] = P.V, O[4] = P.ones (row sums)
#pragma unroll
  for (int r = 0; r < 8; ++r) m_i[r] = -3.0e38f;

  FILL_TILES(0, 0);

  for (int w = 0; w < NWIN; ++w) {
    const int cur = w & 1;
    if (w + 1 < NWIN) {
      FILL_TILES(1 - cur, (w + 1) * 32);       // prev trailing barrier protects this buffer
      WAIT_ASYNC(2);                           // drain current tile; next stays in flight
    } else {
      WAIT_ASYNC(0);
    }
    __syncthreads();                           // whole tile visible to all waves

    // S[jj] = Q . K^T from LDS K tile
    v8f S[2] = {};
#pragma unroll
    for (int jj = 0; jj < 2; ++jj) {
#pragma unroll
      for (int kk = 0; kk < 2; ++kk) {
        HV16 b;  // lane n = key index (row jj*16+lm), contiguous 16 K features
        b.v = *(const v16h*)(&kbuf[cur][jj * 16 + lm][kk * 32 + hiK * 16]);
        S[jj] = WMMA_F16(qa[kk].v, b.v, S[jj]);
      }
    }

    // Online softmax: DPP row-max, exp, rescale (row = vgpr + 8*hiK in C layout)
    float p0[8], p1[8], al[8];
#pragma unroll
    for (int r = 0; r < 8; ++r) {
      const float mx = row16_max(fmaxf(S[0][r], S[1][r]));
      const float mn = fmaxf(m_i[r], mx);
      al[r] = __expf(m_i[r] - mn);
      m_i[r] = mn;
      p0[r] = __expf(S[0][r] - mn);
      p1[r] = __expf(S[1][r] - mn);
    }
#pragma unroll
    for (int nn = 0; nn < 5; ++nn)
#pragma unroll
      for (int r = 0; r < 8; ++r) O[nn][r] *= al[r];

    // C-layout P -> LDS (16x32 f16) -> reload in A layout
#pragma unroll
    for (int r = 0; r < 8; ++r) {
      const int row = r + hiK * 8;
      plds[wave][row][lm]      = (_Float16)p0[r];
      plds[wave][row][16 + lm] = (_Float16)p1[r];
    }
    asm volatile("s_wait_dscnt 0" ::: "memory");
    HV16 pa;
    pa.h[0] = *(const v8h*)(&plds[wave][lm][hiK * 16]);
    pa.h[1] = *(const v8h*)(&plds[wave][lm][hiK * 16 + 8]);

    // O += P . V from LDS V tile (lane n = dim, contiguous 16 keys)
#pragma unroll
    for (int nn = 0; nn < 4; ++nn) {
      HV16 b;
      b.v = *(const v16h*)(&vbuf[cur][nn * 16 + lm][hiK * 16]);
      O[nn] = WMMA_F16(pa.v, b.v, O[nn]);
    }
    O[4] = WMMA_F16(pa.v, bones.v, O[4]);      // row-sum column (softmax denominator)

    __syncthreads();                           // all waves done with buf[cur]
  }

  // Denominator from O[4] (only column 0 non-zero -> row16_sum = row sum), normalize, store
  const int b = bh >> 4, head = bh & 15;
  float inv[8];
#pragma unroll
  for (int r = 0; r < 8; ++r) inv[r] = 1.0f / row16_sum(O[4][r]);
#pragma unroll
  for (int nn = 0; nn < 4; ++nn)
#pragma unroll
    for (int r = 0; r < 8; ++r) {
      const int row = r + hiK * 8;
      const float val = O[nn][r] * inv[r];
      const int s = qt * 16 + row;
      Aout[((size_t)(b * SEQ + s)) * EMB + head * HDIM + nn * 16 + lm] = (_Float16)val;
    }
#undef FILL_TILES
}

// ---------------------------------------------------------------------------
// Launcher
// ---------------------------------------------------------------------------
extern "C" void kernel_launch(void* const* d_in, const int* in_sizes, int n_in,
                              void* d_out, int out_size, void* d_ws, size_t ws_size,
                              hipStream_t stream) {
  const float* query = (const float*)d_in[0];
  const float* key   = (const float*)d_in[1];
  const float* value = (const float*)d_in[2];
  const float* Wq = (const float*)d_in[3];
  const float* bq = (const float*)d_in[4];
  const float* Wk = (const float*)d_in[5];
  const float* bk = (const float*)d_in[6];
  const float* Wv = (const float*)d_in[7];
  const float* bv = (const float*)d_in[8];
  const float* Wo = (const float*)d_in[9];
  const float* bo = (const float*)d_in[10];
  float* out = (float*)d_out;

  char* ws = (char*)d_ws;
  size_t off = 0;
  auto alloc = [&](size_t bytes) -> void* {
    void* p = ws + off;
    off = (off + bytes + 255) & ~(size_t)255;
    return p;
  };
  const size_t nTE = (size_t)T_TOK * EMB;     // 4,718,592
  const size_t nEE = (size_t)EMB * EMB;       // 1,048,576

  _Float16* Xq  = (_Float16*)alloc(nTE * 2);
  _Float16* Xk  = (_Float16*)alloc(nTE * 2);
  _Float16* Xv  = (_Float16*)alloc(nTE * 2);
  _Float16* Wqh = (_Float16*)alloc(nEE * 2);
  _Float16* Wkh = (_Float16*)alloc(nEE * 2);
  _Float16* Wvh = (_Float16*)alloc(nEE * 2);
  _Float16* Woh = (_Float16*)alloc(nEE * 2);
  _Float16* Qh  = (_Float16*)alloc((size_t)BHEADS * SEQ * HDIM * 2);
  _Float16* Khh = (_Float16*)alloc((size_t)BHEADS * SEQ * HDIM * 2);
  _Float16* Vth = (_Float16*)alloc((size_t)BHEADS * HDIM * SEQ * 2);
  _Float16* Att = (_Float16*)alloc(nTE * 2);
  float* sq = (float*)alloc(48 * 16 * 4);
  float* cq = (float*)alloc(48 * 16 * 4);
  float* sk = (float*)alloc(48 * 16 * 4);
  float* ck = (float*)alloc(48 * 16 * 4);

  // 1) converts + tables
  f32_to_f16_kernel<<<(int)((nTE + 255) / 256), 256, 0, stream>>>(query, Xq, (int)nTE);
  f32_to_f16_kernel<<<(int)((nTE + 255) / 256), 256, 0, stream>>>(key,   Xk, (int)nTE);
  f32_to_f16_kernel<<<(int)((nTE + 255) / 256), 256, 0, stream>>>(value, Xv, (int)nTE);
  f32_to_f16_kernel<<<(int)((nEE + 255) / 256), 256, 0, stream>>>(Wq, Wqh, (int)nEE);
  f32_to_f16_kernel<<<(int)((nEE + 255) / 256), 256, 0, stream>>>(Wk, Wkh, (int)nEE);
  f32_to_f16_kernel<<<(int)((nEE + 255) / 256), 256, 0, stream>>>(Wv, Wvh, (int)nEE);
  f32_to_f16_kernel<<<(int)((nEE + 255) / 256), 256, 0, stream>>>(Wo, Woh, (int)nEE);
  xpos_tables_kernel<<<3, 256, 0, stream>>>(sq, cq, sk, ck);

  // 2) projections with fused scale/xPos/head-split (4608 wave-tasks -> 576 blocks)
  proj_gemm_kernel<1><<<576, 256, 0, stream>>>(Xq, Wqh, bq, nullptr, Qh,  nullptr, sq, cq);
  proj_gemm_kernel<2><<<576, 256, 0, stream>>>(Xk, Wkh, bk, nullptr, Khh, nullptr, sk, ck);
  proj_gemm_kernel<3><<<576, 256, 0, stream>>>(Xv, Wvh, bv, nullptr, nullptr, Vth, nullptr, nullptr);

  // 3) flash attention: 32 heads x 18 query slabs -> 576 blocks
  attention_kernel<<<576, 256, 0, stream>>>(Qh, Khh, Vth, Att);

  // 4) output projection -> d_out (f32)
  proj_gemm_kernel<0><<<576, 256, 0, stream>>>(Att, Woh, bo, out, nullptr, nullptr, nullptr, nullptr);
}
